// LlamaRACEModel_23519240913212
// MI455X (gfx1250) — matrix operand order repaired and checked by reading, not compiled, not run on hardware
//
#include <hip/hip_runtime.h>
#include <hip/hip_bf16.h>
#include <math.h>

// ---------------------------------------------------------------------------
// Model constants
// ---------------------------------------------------------------------------
#define T_SEQ   1024
#define DMODEL  768
#define NHEAD   12
#define DK_     64
#define NLAYER  2
#define HID_    3072
#define VSZ     32000
#define MCOPY   2
#define NSEQ    (MCOPY * NHEAD)        // 24
#define NCHUNK  32
#define CT      32                     // chunk length (NCHUNK*CT == T_SEQ)
#define RACE_EPS 1e-6f
#define RMS_EPS  1e-5f

typedef __attribute__((ext_vector_type(16))) __bf16         v16bf;
typedef __attribute__((ext_vector_type(8)))  float          v8f;
typedef __attribute__((ext_vector_type(8)))  unsigned short v8us;
typedef __attribute__((ext_vector_type(16))) unsigned short v16us;

// pack two f32 -> two bf16 (round-half-up) in 3 VALU: add, add, v_perm_b32
__device__ __forceinline__ unsigned pack2bf(float lo, float hi) {
    const unsigned ul = __builtin_bit_cast(unsigned, lo) + 0x8000u;
    const unsigned uh = __builtin_bit_cast(unsigned, hi) + 0x8000u;
    return __builtin_amdgcn_perm(uh, ul, 0x07060302);   // {uh[31:16], ul[31:16]}
}

// A fragment: row-major LDS row pointer; elems 0..7 = K klo..klo+7, 8..15 = K klo+16..klo+23
__device__ __forceinline__ v16bf load_frag_a(const unsigned short* p, int klo) {
    v8us lo = *(const v8us*)(p + klo);
    v8us hi = *(const v8us*)(p + klo + 16);
    v16us t;
    for (int i = 0; i < 8; ++i) { t[i] = lo[i]; t[i + 8] = hi[i]; }
    return __builtin_bit_cast(v16bf, t);
}
// B fragment: col-major LDS row pointer (Bs[col][k]); elems 0..7 = K klo..klo+7, 8..15 = K klo+16..klo+23
__device__ __forceinline__ v16bf load_frag_b(const unsigned short* p, int klo) {
    v8us lo = *(const v8us*)(p + klo);
    v8us hi = *(const v8us*)(p + klo + 16);
    v16us t;
    for (int i = 0; i < 8; ++i) { t[i] = lo[i]; t[i + 8] = hi[i]; }
    return __builtin_bit_cast(v16bf, t);
}

// ---------------------------------------------------------------------------
// WMMA GEMM: C[M,N] = A[M,K] @ B[K,N] (+bias[N]) (+res[M,N])
// BM=128 BN=64 BK=32, 256 threads = 8 waves (4 x 2), each wave 32x32 output.
// Software-pipelined double buffering: global loads for tile k+1 issue as a
// clause BEFORE the WMMAs of tile k; pack+ds_store (which waits loadcnt)
// happens AFTER them, so global latency hides behind matrix math.
// ---------------------------------------------------------------------------
#define BM 128
#define BN 64
#define BK 32
#define LSTR 40   // bf16 stride padding -> conflict-free b128 fragment reads

struct StageRegs { float4 a[4]; float4 b0, b1; };

__device__ __forceinline__ StageRegs load_tile_regs(
    const float* __restrict__ A, const float* __restrict__ B,
    int N, int K, int bm0, int bn0, int k0,
    int ar, int ac, int bp, int bc)
{
    StageRegs r;
    for (int i = 0; i < 4; ++i)
        r.a[i] = *(const float4*)(A + (size_t)(bm0 + ar + i * 32) * K + k0 + ac);
    const float* b0 = B + (size_t)(k0 + 2 * bp) * N + bn0 + bc;
    r.b0 = *(const float4*)(b0);
    r.b1 = *(const float4*)(b0 + N);
    return r;
}

__device__ __forceinline__ void pack_store(
    const StageRegs& r, unsigned short* As, unsigned short* Bs,
    int ar, int ac, int bp, int bc)
{
    for (int i = 0; i < 4; ++i) {
        uint2 w;
        w.x = pack2bf(r.a[i].x, r.a[i].y);
        w.y = pack2bf(r.a[i].z, r.a[i].w);
        *(uint2*)(&As[(ar + i * 32) * LSTR + ac]) = w;
    }
    *(unsigned*)(&Bs[(bc + 0) * LSTR + 2 * bp]) = pack2bf(r.b0.x, r.b1.x);
    *(unsigned*)(&Bs[(bc + 1) * LSTR + 2 * bp]) = pack2bf(r.b0.y, r.b1.y);
    *(unsigned*)(&Bs[(bc + 2) * LSTR + 2 * bp]) = pack2bf(r.b0.z, r.b1.z);
    *(unsigned*)(&Bs[(bc + 3) * LSTR + 2 * bp]) = pack2bf(r.b0.w, r.b1.w);
}

__device__ __forceinline__ void compute_tile(
    const unsigned short* As, const unsigned short* Bs,
    v8f (&acc)[2][2], int wm, int wn, int row15, int kloA, int kloB)
{
    v16bf af[2], bf[2];
    for (int i = 0; i < 2; ++i) {
        af[i] = load_frag_a(&As[(wm * 32 + i * 16 + row15) * LSTR], kloA);
        bf[i] = load_frag_b(&Bs[(wn * 32 + i * 16 + row15) * LSTR], kloB);
    }
    for (int i = 0; i < 2; ++i)
        for (int j = 0; j < 2; ++j)
            acc[i][j] = __builtin_amdgcn_wmma_f32_16x16x32_bf16(
                false, af[i], false, bf[j], (short)0, acc[i][j], false, false);
}

template<bool BIAS, bool RES, bool NT>
__global__ __launch_bounds__(256) void wmma_gemm(
    const float* __restrict__ A, const float* __restrict__ B,
    float* __restrict__ C,
    const float* __restrict__ bias, const float* __restrict__ res,
    int M, int N, int K)   // K must be a multiple of 2*BK
{
    __shared__ __align__(16) unsigned short As0[BM * LSTR];
    __shared__ __align__(16) unsigned short Bs0[BN * LSTR];
    __shared__ __align__(16) unsigned short As1[BM * LSTR];
    __shared__ __align__(16) unsigned short Bs1[BN * LSTR];

    const int tid  = threadIdx.x;
    const int lane = tid & 31;
    const int wave = tid >> 5;
    const int wm   = wave >> 1;     // 0..3
    const int wn   = wave & 1;      // 0..1
    const int bm0  = blockIdx.y * BM;
    const int bn0  = blockIdx.x * BN;
    const int row15 = lane & 15;
    const int kloA  = (lane & 16) ? 8  : 0;
    const int kloB  = (lane & 16) ? 8  : 0;

    // staging coords
    const int ac = (tid & 7) * 4;         // A: 4 cols (k) per thread
    const int ar = (tid >> 3);            // A: base row, +32*i
    const int bp = tid >> 4;              // B: k-pair 0..15
    const int bc = (tid & 15) * 4;        // B: 4 cols per thread

    v8f acc[2][2] = {};

    {   // prologue: stage tile 0 into buffer 0
        StageRegs r0 = load_tile_regs(A, B, N, K, bm0, bn0, 0, ar, ac, bp, bc);
        pack_store(r0, As0, Bs0, ar, ac, bp, bc);
    }

    for (int k0 = 0; k0 < K; k0 += 2 * BK) {
        __syncthreads();
        {   // loads for tile k0+BK issue before WMMAs of tile k0
            StageRegs r1 = load_tile_regs(A, B, N, K, bm0, bn0, k0 + BK, ar, ac, bp, bc);
            compute_tile(As0, Bs0, acc, wm, wn, row15, kloA, kloB);
            pack_store(r1, As1, Bs1, ar, ac, bp, bc);
        }
        __syncthreads();
        if (k0 + 2 * BK < K) {
            StageRegs r0 = load_tile_regs(A, B, N, K, bm0, bn0, k0 + 2 * BK, ar, ac, bp, bc);
            compute_tile(As1, Bs1, acc, wm, wn, row15, kloA, kloB);
            pack_store(r0, As0, Bs0, ar, ac, bp, bc);
        } else {
            compute_tile(As1, Bs1, acc, wm, wn, row15, kloA, kloB);
        }
    }

    // epilogue: VGPR e -> row e + 8*(lane>=16), col = lane&15
    const int rb = (lane >> 4) * 8;
    for (int i = 0; i < 2; ++i)
        for (int j = 0; j < 2; ++j) {
            const int col = bn0 + wn * 32 + j * 16 + row15;
            float bv = 0.0f;
            if (BIAS) bv = bias[col];
            for (int e = 0; e < 8; ++e) {
                const int row = bm0 + wm * 32 + i * 16 + rb + e;
                const size_t idx = (size_t)row * N + col;
                float v = acc[i][j][e] + bv;
                if (RES) v += res[idx];
                if (NT) __builtin_nontemporal_store(v, &C[idx]);
                else    C[idx] = v;
            }
        }
}

// ---------------------------------------------------------------------------
// Elementwise / small kernels
// ---------------------------------------------------------------------------
__global__ void gather_kernel(const int* __restrict__ idx,
                              const float* __restrict__ emb,
                              float* __restrict__ x)
{
    const int t = blockIdx.x;
    const int row = idx[t];
    for (int i = threadIdx.x; i < DMODEL; i += 256)
        x[(size_t)t * DMODEL + i] = emb[(size_t)row * DMODEL + i];
}

__global__ void rmsnorm_kernel(const float* __restrict__ x,
                               const float* __restrict__ w,
                               float* __restrict__ out)
{
    __shared__ float red[256];
    const int t = blockIdx.x;
    const float* xr = x + (size_t)t * DMODEL;
    float s = 0.0f;
    for (int i = threadIdx.x; i < DMODEL; i += 256) { float v = xr[i]; s += v * v; }
    red[threadIdx.x] = s;
    __syncthreads();
    for (int off = 128; off > 0; off >>= 1) {
        if (threadIdx.x < off) red[threadIdx.x] += red[threadIdx.x + off];
        __syncthreads();
    }
    const float rs = rsqrtf(red[0] / (float)DMODEL + RMS_EPS);
    for (int i = threadIdx.x; i < DMODEL; i += 256)
        out[(size_t)t * DMODEL + i] = xr[i] * rs * w[i];
}

// RoPE in-place on (T, H*DK): out[i] = e*cos - o*sin ; out[32+i] = e*sin + o*cos
__global__ void rope_kernel(float* __restrict__ Z)
{
    __shared__ float rowbuf[DMODEL];
    const int t = blockIdx.x;
    for (int i = threadIdx.x; i < DMODEL; i += 256) rowbuf[i] = Z[(size_t)t * DMODEL + i];
    __syncthreads();
    for (int it = threadIdx.x; it < NHEAD * 32; it += 256) {
        const int h = it >> 5, i = it & 31;
        const float theta = __expf(-(2.0f * (float)i / 64.0f) * 9.210340372f); // ln(10000)
        const float f = (float)t * theta;
        const float sn = __sinf(f), cs = __cosf(f);
        const float e = rowbuf[h * 64 + 2 * i];
        const float o = rowbuf[h * 64 + 2 * i + 1];
        Z[(size_t)t * DMODEL + h * 64 + i]      = e * cs - o * sn;
        Z[(size_t)t * DMODEL + h * 64 + 32 + i] = e * sn + o * cs;
    }
}

__global__ void mean_kernel(const float* __restrict__ out2, float* __restrict__ om)
{
    const int i = blockIdx.x * 256 + threadIdx.x;   // 786432 elems
    om[i] = 0.5f * (out2[i] + out2[i + NHEAD * T_SEQ * DK_]);
}

__global__ void silu_mul_kernel(float* __restrict__ u, const float* __restrict__ vv)
{
    const int i = blockIdx.x * 256 + threadIdx.x;   // T*HID elems
    const float x = u[i];
    u[i] = (x / (1.0f + __expf(-x))) * vv[i];
}

// ---------------------------------------------------------------------------
// RACE attention, pass 1: probsK / probsQ for every (n, t) — fully parallel.
// block = 64 threads, grid = NSEQ * T_SEQ.
// ---------------------------------------------------------------------------
__global__ __launch_bounds__(64) void race_probs_kernel(
    const float* __restrict__ Qb, const float* __restrict__ Kb,
    const float* __restrict__ planes,   // [MCOPY][64][16] for this layer
    float* __restrict__ pk, float* __restrict__ pq)
{
    __shared__ float kvec[64], qvec[64], th[32], lg[128];
    const int bid = blockIdx.x;
    const int n = bid >> 10;
    const int t = bid & 1023;
    const int m = n / NHEAD, hh = n % NHEAD;
    const int tid = threadIdx.x;

    kvec[tid] = Kb[(size_t)t * DMODEL + hh * 64 + tid];
    qvec[tid] = Qb[(size_t)t * DMODEL + hh * 64 + tid];
    __syncthreads();

    if (tid < 32) {
        const int mat = tid >> 4, s = tid & 15;
        const float* vec = mat ? qvec : kvec;
        const float* pl = planes + (size_t)m * 64 * 16 + s;
        float acc = 0.0f;
        for (int d = 0; d < 64; ++d) acc += vec[d] * pl[d * 16];
        th[tid] = tanhf(acc) * 0.125f;    // /sqrt(DK)
    }
    __syncthreads();

    for (int q = 0; q < 2; ++q) {         // 128 logits, 2 per thread
        const int id = q * 64 + tid;
        const int mat = id >> 6, l = (id >> 4) & 3, r = id & 15;
        float acc = 0.0f;
        for (int b = 0; b < 4; ++b) {
            const float sgn = ((r >> (3 - b)) & 1) ? 1.0f : -1.0f;
            acc += th[mat * 16 + l * 4 + b] * sgn;
        }
        lg[id] = acc;
    }
    __syncthreads();

    for (int q = 0; q < 2; ++q) {
        const int id = q * 64 + tid;
        const int base = id & ~15;
        float mx = -1e30f;
        for (int r = 0; r < 16; ++r) mx = fmaxf(mx, lg[base + r]);
        float sum = 0.0f;
        for (int r = 0; r < 16; ++r) sum += __expf(lg[base + r] - mx);
        const float p = __expf(lg[id] - mx) / sum;
        float* outp = (id < 64) ? pk : pq;
        outp[(size_t)(n * T_SEQ + t) * 64 + (id & 63)] = p;
    }
}

// ---------------------------------------------------------------------------
// Pass 2: per-chunk totals of sum(p) and sum(p (x) v). grid = NSEQ*NCHUNK.
// ---------------------------------------------------------------------------
__global__ __launch_bounds__(256) void race_chunk_totals(
    const float* __restrict__ pk, const float* __restrict__ Vb,
    float* __restrict__ totS, float* __restrict__ totA)
{
    const int bid = blockIdx.x;
    const int n = bid >> 5, c = bid & 31;
    const int hh = n % NHEAD;
    const int tid = threadIdx.x;
    const int lr = tid & 63, db = tid >> 6;

    __shared__ float pkr[64], vr[64];
    float acc[16];
    for (int j = 0; j < 16; ++j) acc[j] = 0.0f;
    float accA = 0.0f;

    for (int tt = 0; tt < CT; ++tt) {
        const int t = c * CT + tt;
        __syncthreads();
        if (tid < 64)        pkr[tid]      = pk[(size_t)(n * T_SEQ + t) * 64 + tid];
        else if (tid < 128)  vr[tid - 64]  = Vb[(size_t)t * DMODEL + hh * 64 + (tid - 64)];
        __syncthreads();
        const float p = pkr[lr];
        for (int j = 0; j < 16; ++j) acc[j] += p * vr[db * 16 + j];
        if (tid < 64) accA += pkr[tid];
    }
    const size_t base = ((size_t)bid * 64 + lr) * 64 + db * 16;
    for (int j = 0; j < 16; ++j) totS[base + j] = acc[j];
    if (tid < 64) totA[(size_t)bid * 64 + tid] = accA;
}

// ---------------------------------------------------------------------------
// Pass 3: exclusive scan of chunk totals over c. grid = NSEQ.
// ---------------------------------------------------------------------------
__global__ __launch_bounds__(256) void race_scan_totals(
    const float* __restrict__ totS, const float* __restrict__ totA,
    float* __restrict__ prefS, float* __restrict__ prefA)
{
    const int n = blockIdx.x;
    const int tid = threadIdx.x;
    float acc[16];
    for (int j = 0; j < 16; ++j) acc[j] = 0.0f;
    for (int c = 0; c < NCHUNK; ++c) {
        const size_t base = (size_t)(n * NCHUNK + c) * 4096 + tid * 16;
        for (int j = 0; j < 16; ++j) { prefS[base + j] = acc[j]; acc[j] += totS[base + j]; }
    }
    if (tid < 64) {
        float a = 0.0f;
        for (int c = 0; c < NCHUNK; ++c) {
            const size_t base = (size_t)(n * NCHUNK + c) * 64 + tid;
            prefA[base] = a; a += totA[base];
        }
    }
}

// ---------------------------------------------------------------------------
// Pass 4: local 32-step inclusive scan + output. grid = NSEQ*NCHUNK.
// State S[64 lr][64 d] in LDS (65-pad), a[64].
// ---------------------------------------------------------------------------
__global__ __launch_bounds__(256) void race_local_scan(
    const float* __restrict__ pk, const float* __restrict__ pq,
    const float* __restrict__ Vb,
    const float* __restrict__ prefS, const float* __restrict__ prefA,
    float* __restrict__ out2)
{
    const int bid = blockIdx.x;
    const int n = bid >> 5, c = bid & 31;
    const int hh = n % NHEAD;
    const int tid = threadIdx.x;
    const int lr = tid & 63, db = tid >> 6;

    __shared__ float S[64 * 65];
    __shared__ float a[64];
    __shared__ float wq[64];
    __shared__ float pkr[64], pqr[64], vr[64];
    __shared__ float part[4][64];

    {
        const size_t base = (size_t)bid * 4096 + tid * 16;
        for (int j = 0; j < 16; ++j) {
            const int flat = tid * 16 + j;
            S[(flat >> 6) * 65 + (flat & 63)] = prefS[base + j];
        }
        if (tid < 64) a[tid] = prefA[(size_t)bid * 64 + tid];
    }

    for (int tt = 0; tt < CT; ++tt) {
        const int t = c * CT + tt;
        __syncthreads();
        if (tid < 64)        pkr[tid]        = pk[(size_t)(n * T_SEQ + t) * 64 + tid];
        else if (tid < 128)  pqr[tid - 64]   = pq[(size_t)(n * T_SEQ + t) * 64 + (tid - 64)];
        else if (tid < 192)  vr[tid - 128]   = Vb[(size_t)t * DMODEL + hh * 64 + (tid - 128)];
        __syncthreads();
        // inclusive update
        {
            const float p = pkr[lr];
            for (int j = 0; j < 16; ++j)
                S[lr * 65 + db * 16 + j] += p * vr[db * 16 + j];
            if (tid < 64) a[tid] += pkr[tid];
        }
        __syncthreads();
        if (tid < 64) wq[tid] = pqr[tid] / (a[tid] + RACE_EPS);
        __syncthreads();
        // out[d] = sum_lr wq[lr] * S[lr][d]  (4-way partial reduction)
        {
            const int d = lr, g = db;
            float s = 0.0f;
            for (int q = 0; q < 16; ++q) {
                const int li = g * 16 + q;
                s += wq[li] * S[li * 65 + d];
            }
            part[g][d] = s;
        }
        __syncthreads();
        if (tid < 64)
            out2[(size_t)(n * T_SEQ + t) * 64 + tid] =
                part[0][tid] + part[1][tid] + part[2][tid] + part[3][tid];
    }
}

// ---------------------------------------------------------------------------
// Host orchestration
// ---------------------------------------------------------------------------
extern "C" void kernel_launch(void* const* d_in, const int* in_sizes, int n_in,
                              void* d_out, int out_size, void* d_ws, size_t ws_size,
                              hipStream_t stream)
{
    (void)in_sizes; (void)n_in; (void)out_size; (void)ws_size;

    const int*   in_idx       = (const int*)  d_in[0];
    const float* tok_emb      = (const float*)d_in[1];
    const float* norm1_w      = (const float*)d_in[2];
    const float* qw           = (const float*)d_in[3];
    const float* kw           = (const float*)d_in[4];
    const float* vw           = (const float*)d_in[5];
    const float* ow           = (const float*)d_in[6];
    const float* ob           = (const float*)d_in[7];
    const float* norm2_w      = (const float*)d_in[8];
    const float* ff_w1        = (const float*)d_in[9];
    const float* ff_w2        = (const float*)d_in[10];
    const float* ff_w3        = (const float*)d_in[11];
    const float* final_norm_w = (const float*)d_in[12];
    const float* out_w        = (const float*)d_in[13];
    const float* planes       = (const float*)d_in[14];

    float* ws = (float*)d_ws;
    size_t off = 0;
    auto alloc = [&](size_t n) { float* p = ws + off; off += (n + 63) & ~(size_t)63; return p; };

    float* x     = alloc((size_t)T_SEQ * DMODEL);
    float* h     = alloc((size_t)T_SEQ * DMODEL);
    float* Qb    = alloc((size_t)T_SEQ * DMODEL);
    float* Kb    = alloc((size_t)T_SEQ * DMODEL);
    float* Vb    = alloc((size_t)T_SEQ * DMODEL);
    float* pk    = alloc((size_t)NSEQ * T_SEQ * DK_);
    float* pq    = alloc((size_t)NSEQ * T_SEQ * DK_);
    float* totS  = alloc((size_t)NSEQ * NCHUNK * 4096);
    float* totA  = alloc((size_t)NSEQ * NCHUNK * 64);
    float* prefS = alloc((size_t)NSEQ * NCHUNK * 4096);
    float* prefA = alloc((size_t)NSEQ * NCHUNK * 64);
    float* out2  = alloc((size_t)NSEQ * T_SEQ * DK_);
    float* omean = Qb;     // Q no longer needed once out2 exists
    float* u     = totS;   // FFN phase reuses scan scratch (3.15M floats each)
    float* vv    = prefS;

    auto gemm = [&](const float* A, const float* B, float* C,
                    const float* bias, const float* res,
                    int M, int N, int K, bool nt) {
        dim3 grid(N / BN, M / BM);
        if (bias && res)
            wmma_gemm<true,  true,  false><<<grid, 256, 0, stream>>>(A, B, C, bias, res, M, N, K);
        else if (res)
            wmma_gemm<false, true,  false><<<grid, 256, 0, stream>>>(A, B, C, bias, res, M, N, K);
        else if (nt)
            wmma_gemm<false, false, true ><<<grid, 256, 0, stream>>>(A, B, C, bias, res, M, N, K);
        else
            wmma_gemm<false, false, false><<<grid, 256, 0, stream>>>(A, B, C, bias, res, M, N, K);
    };

    gather_kernel<<<T_SEQ, 256, 0, stream>>>(in_idx, tok_emb, x);

    for (int l = 0; l < NLAYER; ++l) {
        const size_t wD2 = (size_t)l * DMODEL * DMODEL;

        rmsnorm_kernel<<<T_SEQ, 256, 0, stream>>>(x, norm1_w + l * DMODEL, h);

        gemm(h, qw + wD2, Qb, nullptr, nullptr, T_SEQ, DMODEL, DMODEL, false);
        gemm(h, kw + wD2, Kb, nullptr, nullptr, T_SEQ, DMODEL, DMODEL, false);
        gemm(h, vw + wD2, Vb, nullptr, nullptr, T_SEQ, DMODEL, DMODEL, false);

        rope_kernel<<<T_SEQ, 256, 0, stream>>>(Qb);
        rope_kernel<<<T_SEQ, 256, 0, stream>>>(Kb);

        race_probs_kernel<<<NSEQ * T_SEQ, 64, 0, stream>>>(
            Qb, Kb, planes + (size_t)l * MCOPY * 64 * 16, pk, pq);
        race_chunk_totals<<<NSEQ * NCHUNK, 256, 0, stream>>>(pk, Vb, totS, totA);
        race_scan_totals<<<NSEQ, 256, 0, stream>>>(totS, totA, prefS, prefA);
        race_local_scan<<<NSEQ * NCHUNK, 256, 0, stream>>>(pk, pq, Vb, prefS, prefA, out2);

        mean_kernel<<<(NHEAD * T_SEQ * DK_) / 256, 256, 0, stream>>>(out2, omean);

        gemm(omean, ow + wD2, x, ob + l * DMODEL, x, T_SEQ, DMODEL, DMODEL, false);

        rmsnorm_kernel<<<T_SEQ, 256, 0, stream>>>(x, norm2_w + l * DMODEL, h);

        const size_t wH = (size_t)l * DMODEL * HID_;
        gemm(h, ff_w1 + wH, u,  nullptr, nullptr, T_SEQ, HID_, DMODEL, false);
        gemm(h, ff_w2 + wH, vv, nullptr, nullptr, T_SEQ, HID_, DMODEL, false);
        silu_mul_kernel<<<((size_t)T_SEQ * HID_) / 256, 256, 0, stream>>>(u, vv);
        gemm(u, ff_w3 + (size_t)l * HID_ * DMODEL, x, nullptr, x, T_SEQ, DMODEL, HID_, false);
    }

    rmsnorm_kernel<<<T_SEQ, 256, 0, stream>>>(x, final_norm_w, h);
    gemm(h, out_w, (float*)d_out, nullptr, nullptr, T_SEQ, VSZ, DMODEL, true);
}
